// GRNN_69818988364143
// MI455X (gfx1250) — compile-verified, matrix operand order checked by reading, not run on hardware
//
#include <hip/hip_runtime.h>

// ---------------------------------------------------------------------------
// GRNN (dual SRU) fused kernel for gfx1250 (MI455X, wave32, WMMA)
//
//   U = x_half @ W  (K=256 -> 3*256 cols)  via v_wmma_f32_16x16x32_f16
//   c_t = f_t*c_{t-1} + (1-f_t)*xt_t ; h = r*c + (1-r)*x   (serial scan)
//
// One workgroup = (batch b, half, 32-channel tile). 3 waves:
//   wave0 -> x_tilde columns, wave1 -> f columns, wave2 -> r columns.
// W slice resident in LDS (f16, 48KB). x streamed in 16-step time chunks.
// ---------------------------------------------------------------------------

typedef __attribute__((ext_vector_type(16))) _Float16 v16h;
typedef __attribute__((ext_vector_type(8)))  float    v8f;

#define B_   32
#define T_   2000
#define DF   512          // full feature dim
#define DH   256          // per-half dim (== H)
#define CT   32           // channels per workgroup
#define TT   16           // time steps per chunk (2000 = 125 * 16)
#define NCH  (T_ / TT)    // 125 chunks
#define KW   256          // GEMM K
#define NW   96           // 3 gates * 32 channels (cols owned by this WG)
#define UP   17           // padded t-stride for Us (bank-conflict free)
#define THREADS 96

union ABH { v16h v; float4 f4[2]; };
struct __align__(8) H4 { _Float16 h[4]; };

__device__ __forceinline__ float sigmoidf_(float z) {
    return 1.0f / (1.0f + __expf(-z));
}

// Stage one 16 x 256 f32 x-tile from global into LDS as f16 (row-major).
__device__ __forceinline__ void load_A_tile(_Float16* __restrict__ As,
                                            const float* __restrict__ xg,
                                            int b, int t0, int xbase,
                                            int tid, int nthr) {
    // 16 rows * 256 cols / 4 = 1024 float4 units
    for (int u = tid; u < (TT * KW) / 4; u += nthr) {
        int row = u >> 6;            // /64
        int kc  = (u & 63) << 2;     // *4
        const float4 v = *(const float4*)(xg + ((size_t)b * T_ + (size_t)(t0 + row)) * DF
                                          + xbase + kc);
        H4 h;
        h.h[0] = (_Float16)v.x; h.h[1] = (_Float16)v.y;
        h.h[2] = (_Float16)v.z; h.h[3] = (_Float16)v.w;
        *(H4*)(&As[row * KW + kc]) = h;
    }
}

__global__ void __launch_bounds__(THREADS)
grnn_sru_fused(const float* __restrict__ x,
               const float* __restrict__ W1, const float* __restrict__ bf1,
               const float* __restrict__ br1,
               const float* __restrict__ W2, const float* __restrict__ bf2,
               const float* __restrict__ br2,
               float* __restrict__ out) {
    __shared__ __align__(16) _Float16 Ws[NW * KW];   // 48 KB: W slice, [col][k], f16
    __shared__ __align__(16) _Float16 As[TT * KW];   //  8 KB: x chunk,  [t][k],  f16
    __shared__ float Us[NW * UP];                    // 6.4 KB: U chunk, [col][t] padded

    const int blk  = blockIdx.x;          // 0..511
    const int b    = blk >> 4;            // batch
    const int half = (blk >> 3) & 1;      // which SRU
    const int ct   = blk & 7;             // 32-channel tile within H

    const float* Wm = half ? W2 : W1;
    const float* bf = half ? bf2 : bf1;
    const float* br = half ? br2 : br1;

    const int tid  = threadIdx.x;
    const int wv   = tid >> 5;            // wave: 0=x_tilde, 1=f, 2=r
    const int lane = tid & 31;

    const int xbase = half * DH;          // K-columns of x for this half

    // ---- stage W slice into LDS (f16, transposed: Ws[col][k]) --------------
    {
        const int g    = tid >> 5;                    // gate group
        const int n    = tid & 31;                    // channel in tile
        const int gcol = g * DH + ct * CT + n;        // column in (256 x 768) W
        for (int k = 0; k < KW; ++k) {
            Ws[tid * KW + k] = (_Float16)Wm[(size_t)k * (3 * DH) + gcol];
        }
    }
    // ---- stage first x tile -------------------------------------------------
    load_A_tile(As, x, b, 0, xbase, tid, THREADS);
    __syncthreads();

    // per-lane constant addressing for WMMA operand fetch (wave32 layout)
    const int arow = lane & 15;
    const int koff = (lane < 16) ? 0 : 8;             // K sub-offset per half-wave
    const _Float16* Arow  = &As[arow * KW];
    const _Float16* Brow0 = &Ws[(wv * 32 + (lane & 15)) * KW];
    const _Float16* Brow1 = &Ws[(wv * 32 + 16 + (lane & 15)) * KW];

    float carry = 0.0f;                               // recurrence state (wave0)
    float bfv = 0.0f, brv = 0.0f;
    if (wv == 0) {
        bfv = bf[ct * CT + lane];
        brv = br[ct * CT + lane];
    }

    for (int ch = 0; ch < NCH; ++ch) {
        // ---- GEMM phase: U[t, cols] for this wave's 2 N-tiles ---------------
        v8f acc0 = {0.f, 0.f, 0.f, 0.f, 0.f, 0.f, 0.f, 0.f};
        v8f acc1 = {0.f, 0.f, 0.f, 0.f, 0.f, 0.f, 0.f, 0.f};
        #pragma unroll
        for (int ks = 0; ks < KW / 32; ++ks) {
            const int kb = ks * 32 + koff;
            ABH a, b0, b1;
            a.f4[0]  = *(const float4*)(Arow  + kb);
            a.f4[1]  = *(const float4*)(Arow  + kb + 16);
            b0.f4[0] = *(const float4*)(Brow0 + kb);
            b0.f4[1] = *(const float4*)(Brow0 + kb + 16);
            b1.f4[0] = *(const float4*)(Brow1 + kb);
            b1.f4[1] = *(const float4*)(Brow1 + kb + 16);
            acc0 = __builtin_amdgcn_wmma_f32_16x16x32_f16(
                       false, a.v, false, b0.v, (short)0, acc0, false, false);
            acc1 = __builtin_amdgcn_wmma_f32_16x16x32_f16(
                       false, a.v, false, b1.v, (short)0, acc1, false, false);
        }
        // ---- spill U tiles to padded LDS (C layout: lane=N, vgpr i -> M) ----
        {
            const int N  = lane & 15;
            const int Mb = (lane >> 4) * 8;
            const int c0 = wv * 32 + N;
            const int c1 = wv * 32 + 16 + N;
            #pragma unroll
            for (int i = 0; i < 8; ++i) {
                Us[c0 * UP + Mb + i] = acc0[i];
                Us[c1 * UP + Mb + i] = acc1[i];
            }
        }
        __syncthreads();

        if (wv != 0) {
            // waves 1,2: prefetch next x tile while wave0 scans
            if (ch + 1 < NCH) {
                load_A_tile(As, x, b, (ch + 1) * TT, xbase, tid - 32, 64);
            }
        } else {
            // wave0: serial recurrence + highway for this chunk (lane = channel)
            const int t0   = ch * TT;
            const int xcol = half * DH + ct * CT + lane;
            const float* xp = x   + ((size_t)b * T_ + t0) * DF + xcol;
            float*       op = out + ((size_t)b * T_ + t0) * DF + xcol;
            float c = carry;
            #pragma unroll
            for (int t = 0; t < TT; ++t) {
                const float xt = Us[(0 * 32 + lane) * UP + t];
                const float fp = Us[(32 + lane) * UP + t];
                const float rp = Us[(64 + lane) * UP + t];
                const float f  = sigmoidf_(fp + bfv);
                const float r  = sigmoidf_(rp + brv);
                c = f * c + (1.0f - f) * xt;
                const float xv = xp[(size_t)t * DF];
                op[(size_t)t * DF] = r * c + (1.0f - r) * xv;
            }
            carry = c;
        }
        __syncthreads();
    }
}

extern "C" void kernel_launch(void* const* d_in, const int* in_sizes, int n_in,
                              void* d_out, int out_size, void* d_ws, size_t ws_size,
                              hipStream_t stream) {
    const float* x   = (const float*)d_in[0];
    const float* W1  = (const float*)d_in[1];
    const float* bf1 = (const float*)d_in[2];
    const float* br1 = (const float*)d_in[3];
    const float* W2  = (const float*)d_in[4];
    const float* bf2 = (const float*)d_in[5];
    const float* br2 = (const float*)d_in[6];
    float* out = (float*)d_out;

    dim3 grid(B_ * 2 * (DH / CT));   // 32 batches * 2 halves * 8 channel tiles = 512
    grnn_sru_fused<<<grid, THREADS, 0, stream>>>(x, W1, bf1, br1, W2, bf2, br2, out);
}
